// AttentionHead_73100343377998
// MI455X (gfx1250) — compile-verified, matrix operand order checked by reading, not run on hardware
//
#include <hip/hip_runtime.h>

// Problem constants (match reference)
#define BB 8
#define TT 2048
#define DD 1024
#define HH 64

typedef __attribute__((ext_vector_type(16))) _Float16 v16h;
typedef __attribute__((ext_vector_type(8)))  _Float16 v8h;
typedef __attribute__((ext_vector_type(8)))  float    v8f;
typedef __attribute__((ext_vector_type(4)))  float    v4f;

#define WMMA_F16(a, b, c) \
    __builtin_amdgcn_wmma_f32_16x16x32_f16(false, (a), false, (b), (short)0, (c), false, false)

// ---- gfx1250 async global->LDS path (guarded; falls back to direct loads) ---
#if __has_builtin(__builtin_amdgcn_global_load_async_to_lds_b128)
#define HAVE_ASYNC_LDS 1
typedef int v4i_vs __attribute__((vector_size(16)));            // matches builtin's V4i
typedef __attribute__((address_space(1))) v4i_vs as1_v4i;       // global pointee
typedef __attribute__((address_space(3))) v4i_vs as3_v4i;       // LDS pointee
#endif

#if __has_builtin(__builtin_amdgcn_s_wait_asynccnt)
#define WAIT_ASYNC(n) __builtin_amdgcn_s_wait_asynccnt(n)
#else
#define WAIT_ASYNC(n) asm volatile("s_wait_asynccnt %0" ::"i"(n) : "memory")
#endif

// ---- fragment loaders -------------------------------------------------------
// A/B fragment layout (16-bit, 16x32 / 32x16): per lane, two contiguous runs of
// 8 f16 at [kbase .. kbase+7] and [kbase+16 .. kbase+23], kbase = (lane<16)?0:8.

__device__ __forceinline__ v16h load_frag_h16(const _Float16* p) {
    v8h lo = *(const v8h*)p;
    v8h hi = *(const v8h*)(p + 16);
    v16h r;
#pragma unroll
    for (int i = 0; i < 8; ++i) { r[i] = lo[i]; r[i + 8] = hi[i]; }
    return r;
}

__device__ __forceinline__ v16h load_frag_f32(const float* p) {
    v4f a = *(const v4f*)p;
    v4f b = *(const v4f*)(p + 4);
    v4f c = *(const v4f*)(p + 16);
    v4f d = *(const v4f*)(p + 20);
    v16h r;
#pragma unroll
    for (int i = 0; i < 4; ++i) {
        r[i]      = (_Float16)a[i];
        r[i + 4]  = (_Float16)b[i];
        r[i + 8]  = (_Float16)c[i];
        r[i + 12] = (_Float16)d[i];
    }
    return r;
}

// xor-lane shuffle via ds_swizzle (group-of-32 mode: offset = xor<<10 | and=0x1f)
template <int XORMASK>
__device__ __forceinline__ float swz_xor(float v) {
    int i = __builtin_amdgcn_ds_swizzle(__float_as_int(v), (XORMASK << 10) | 0x1f);
    return __int_as_float(i);
}

// ---- kernel 0: W (DxH, f32) -> W^T (HxD, f16), optional scale ---------------
__global__ void wt_convert_kernel(const float* __restrict__ W,
                                  _Float16* __restrict__ Wt, float scale) {
    int idx = blockIdx.x * blockDim.x + threadIdx.x;
    if (idx >= DD * HH) return;
    int h = idx / DD;
    int d = idx - h * DD;
    Wt[h * DD + d] = (_Float16)(W[d * HH + h] * scale);
}

// ---- kernel 1: fused QKV projection (one wave per 16 token rows) ------------
// Q16,K16: [B][T][H] f16 row-major.  Vt16: [B][H][T] f16 (transposed).
__global__ __launch_bounds__(32) void qkv_proj_kernel(
    const float* __restrict__ x,
    const _Float16* __restrict__ Wtq, const _Float16* __restrict__ Wtk,
    const _Float16* __restrict__ Wtv,
    _Float16* __restrict__ Q16, _Float16* __restrict__ K16,
    _Float16* __restrict__ Vt16) {
    const int tiles_per_b = TT / 16;
    const int b  = blockIdx.x / tiles_per_b;
    const int t0 = (blockIdx.x % tiles_per_b) * 16;

    const int lane  = threadIdx.x;
    const int lr    = lane & 15;
    const int hb    = lane >> 4;
    const int kbase = hb ? 8 : 0;

    v8f accq[4], acck[4], accv[4];
#pragma unroll
    for (int nt = 0; nt < 4; ++nt) { accq[nt] = {}; acck[nt] = {}; accv[nt] = {}; }

    const float* xrow = x + ((size_t)(b * TT + t0 + lr)) * DD;

#pragma unroll 1
    for (int k0 = 0; k0 < DD; k0 += 32) {
        v16h af = load_frag_f32(xrow + k0 + kbase);
#pragma unroll
        for (int nt = 0; nt < 4; ++nt) {
            const int n = nt * 16 + lr;
            v16h bq = load_frag_h16(Wtq + (size_t)n * DD + k0 + kbase);
            accq[nt] = WMMA_F16(af, bq, accq[nt]);
            v16h bk = load_frag_h16(Wtk + (size_t)n * DD + k0 + kbase);
            acck[nt] = WMMA_F16(af, bk, acck[nt]);
            v16h bv = load_frag_h16(Wtv + (size_t)n * DD + k0 + kbase);
            accv[nt] = WMMA_F16(af, bv, accv[nt]);
        }
    }

    // C layout: VGPR r, lane -> (M = r + 8*hb, N = lr + 16*nt)
#pragma unroll
    for (int nt = 0; nt < 4; ++nt) {
#pragma unroll
        for (int r = 0; r < 8; ++r) {
            const int t = t0 + r + 8 * hb;
            const int n = nt * 16 + lr;
            Q16[((size_t)b * TT + t) * HH + n] = (_Float16)accq[nt][r];
            K16[((size_t)b * TT + t) * HH + n] = (_Float16)acck[nt][r];
            Vt16[(size_t)b * HH * TT + (size_t)n * TT + t] = (_Float16)accv[nt][r];
        }
    }
}

// ---- kernel 2: streaming causal flash attention (one wave per q-tile) -------
// K/V tiles double-buffered in LDS via gfx1250 async global->LDS DMA when
// available; softmax done in-register with ds_swizzle xor reductions.
__global__ __launch_bounds__(32) void flash_attn_kernel(
    const _Float16* __restrict__ Q16, const _Float16* __restrict__ K16,
    const _Float16* __restrict__ Vt16, float* __restrict__ out) {
    const int tiles_per_b = TT / 16;
    const int b  = blockIdx.x / tiles_per_b;
    const int t0 = (blockIdx.x % tiles_per_b) * 16;

    const int lane  = threadIdx.x;
    const int lr    = lane & 15;
    const int hb    = lane >> 4;
    const int kbase = hb ? 8 : 0;

    __shared__ alignas(16) _Float16 Plds[16 * 32];
#ifdef HAVE_ASYNC_LDS
    __shared__ alignas(16) _Float16 Kls[2][32 * 64];  // [buf][krow][h]   4KB ea
    __shared__ alignas(16) _Float16 Vls[2][64 * 32];  // [buf][h][t-kj]   4KB ea
#endif

    // Q A-fragments over H=64 (two K-chunks of 32); 1/sqrt(H) folded into Wq
    const _Float16* qrow = Q16 + ((size_t)(b * TT + t0 + lr)) * HH;
    v16h aq0 = load_frag_h16(qrow + 0  + kbase);
    v16h aq1 = load_frag_h16(qrow + 32 + kbase);

    v8f o0 = {}, o1 = {}, o2 = {}, o3 = {};
    float mrun[8], lrun[8];
#pragma unroll
    for (int r = 0; r < 8; ++r) { mrun[r] = -1e30f; lrun[r] = 0.0f; }

    const _Float16* kmat  = K16 + (size_t)b * TT * HH;       // [t][h]
    const _Float16* vmat  = Vt16 + (size_t)b * HH * TT;      // [h][t]
    const int kend = t0 + 16;  // causal: need k <= t0+15

#ifdef HAVE_ASYNC_LDS
    // issue 16 async b128 loads covering one 32-token K tile + V tile
    auto issue_tile = [&](int kj, int buf) {
        // K tile: rows kj..kj+31 are contiguous (32*64 halves = 4KB flat copy)
        const _Float16* kg = kmat + (size_t)kj * HH + lane * 8;
        _Float16*       kl = &Kls[buf][lane * 8];
#pragma unroll
        for (int i = 0; i < 8; ++i)
            __builtin_amdgcn_global_load_async_to_lds_b128(
                (as1_v4i*)(kg + i * 256), (as3_v4i*)(kl + i * 256), 0, 0);
        // V tile: 64 rows (h), 32 halves each; global row stride = T halves
        const int vrow = lane >> 2;          // 0..7 (8 rows / instruction)
        const int vcol = (lane & 3) * 8;     // 0,8,16,24
        const _Float16* vg = vmat + (size_t)vrow * TT + kj + vcol;
        _Float16*       vl = &Vls[buf][vrow * 32 + vcol];
#pragma unroll
        for (int i = 0; i < 8; ++i)
            __builtin_amdgcn_global_load_async_to_lds_b128(
                (as1_v4i*)(vg + (size_t)8 * TT * i), (as3_v4i*)(vl + 8 * 32 * i),
                0, 0);
    };
    issue_tile(0, 0);
#endif

#pragma unroll 1
    for (int kj = 0; kj < kend; kj += 32) {
#ifdef HAVE_ASYNC_LDS
        const int cur = (kj >> 5) & 1;
        if (kj + 32 < kend) {
            issue_tile(kj + 32, cur ^ 1);  // prefetch next tile via async DMA
            WAIT_ASYNC(16);                // in-order: current tile resident
        } else {
            WAIT_ASYNC(0);
        }
        const _Float16* ktile = &Kls[cur][0];   // [row][64]
        const _Float16* vtile = &Vls[cur][0];   // [h][32]
        const int krow_stride = HH;   // 64
#else
        if (kj + 32 < kend) {  // prefetch next K/V tiles (global_prefetch_b8)
            __builtin_prefetch(kmat + (size_t)(kj + 32 + lr) * HH, 0, 3);
            __builtin_prefetch(vmat + (size_t)lr * TT + kj + 32, 0, 3);
        }
        const _Float16* ktile = kmat + (size_t)kj * HH;
        const _Float16* vtile = nullptr;  // addressed directly below
        const int krow_stride = HH;
#endif

        // scores: two 16x16 tiles, each accumulating over H with 2 WMMAs
        v8f s0 = {}, s1 = {};
        {
            const _Float16* kr0 = ktile + (size_t)lr * krow_stride;
            s0 = WMMA_F16(aq0, load_frag_h16(kr0 + 0  + kbase), s0);
            s0 = WMMA_F16(aq1, load_frag_h16(kr0 + 32 + kbase), s0);
            const _Float16* kr1 = ktile + (size_t)(16 + lr) * krow_stride;
            s1 = WMMA_F16(aq0, load_frag_h16(kr1 + 0  + kbase), s1);
            s1 = WMMA_F16(aq1, load_frag_h16(kr1 + 32 + kbase), s1);
        }

        // causal mask + online softmax update (C layout: row = r+8*hb, col = lr)
#pragma unroll
        for (int r = 0; r < 8; ++r) {
            const int rg = t0 + r + 8 * hb;
            if (kj + lr      > rg) s0[r] = -1e30f;
            if (kj + 16 + lr > rg) s1[r] = -1e30f;

            float v = fmaxf(s0[r], s1[r]);
            v = fmaxf(v, swz_xor<1>(v));
            v = fmaxf(v, swz_xor<2>(v));
            v = fmaxf(v, swz_xor<4>(v));
            v = fmaxf(v, swz_xor<8>(v));
            const float mn   = fmaxf(mrun[r], v);
            const float corr = __expf(mrun[r] - mn);
            mrun[r] = mn;

            const float p0 = __expf(s0[r] - mn);
            const float p1 = __expf(s1[r] - mn);
            s0[r] = p0; s1[r] = p1;
            float rs = p0 + p1;
            rs += swz_xor<1>(rs);
            rs += swz_xor<2>(rs);
            rs += swz_xor<4>(rs);
            rs += swz_xor<8>(rs);
            lrun[r] = lrun[r] * corr + rs;

            o0[r] *= corr; o1[r] *= corr; o2[r] *= corr; o3[r] *= corr;
        }

        // P (C layout, f32) -> LDS 16x32 row-major f16, re-read as A fragment
#pragma unroll
        for (int r = 0; r < 8; ++r) {
            const int M = r + 8 * hb;
            Plds[M * 32 + lr]      = (_Float16)s0[r];
            Plds[M * 32 + 16 + lr] = (_Float16)s1[r];
        }
        asm volatile("s_wait_dscnt 0" ::: "memory");
        v16h ap = load_frag_h16(&Plds[lr * 32 + kbase]);

        // O += P @ V  (V^T rows are natural B fragments)
#ifdef HAVE_ASYNC_LDS
        o0 = WMMA_F16(ap, load_frag_h16(vtile + (0 * 16 + lr) * 32 + kbase), o0);
        o1 = WMMA_F16(ap, load_frag_h16(vtile + (1 * 16 + lr) * 32 + kbase), o1);
        o2 = WMMA_F16(ap, load_frag_h16(vtile + (2 * 16 + lr) * 32 + kbase), o2);
        o3 = WMMA_F16(ap, load_frag_h16(vtile + (3 * 16 + lr) * 32 + kbase), o3);
#else
        const _Float16* vb = vmat + (size_t)lr * TT + kj + kbase;
        o0 = WMMA_F16(ap, load_frag_h16(vb + (size_t)0  * TT), o0);
        o1 = WMMA_F16(ap, load_frag_h16(vb + (size_t)16 * TT), o1);
        o2 = WMMA_F16(ap, load_frag_h16(vb + (size_t)32 * TT), o2);
        o3 = WMMA_F16(ap, load_frag_h16(vb + (size_t)48 * TT), o3);
#endif
    }

    // normalize + store fp32 output [B][T][H]
#pragma unroll
    for (int r = 0; r < 8; ++r) {
        const float inv = 1.0f / lrun[r];
        const int   t   = t0 + r + 8 * hb;
        float* orow = out + ((size_t)b * TT + t) * HH;
        orow[lr +  0] = o0[r] * inv;
        orow[lr + 16] = o1[r] * inv;
        orow[lr + 32] = o2[r] * inv;
        orow[lr + 48] = o3[r] * inv;
    }
}

// ---- host-side launch -------------------------------------------------------
extern "C" void kernel_launch(void* const* d_in, const int* in_sizes, int n_in,
                              void* d_out, int out_size, void* d_ws, size_t ws_size,
                              hipStream_t stream) {
    const float* x  = (const float*)d_in[0];
    const float* Wq = (const float*)d_in[1];
    const float* Wk = (const float*)d_in[2];
    const float* Wv = (const float*)d_in[3];
    float* out = (float*)d_out;

    // workspace carve-up (f16 buffers)
    char* ws = (char*)d_ws;
    const size_t wt_bytes = (size_t)HH * DD * sizeof(_Float16);      // 128 KB each
    const size_t qk_bytes = (size_t)BB * TT * HH * sizeof(_Float16); // 2 MB each
    _Float16* Wtq  = (_Float16*)(ws);
    _Float16* Wtk  = (_Float16*)(ws + wt_bytes);
    _Float16* Wtv  = (_Float16*)(ws + 2 * wt_bytes);
    _Float16* Q16  = (_Float16*)(ws + 3 * wt_bytes);
    _Float16* K16  = (_Float16*)(ws + 3 * wt_bytes + qk_bytes);
    _Float16* Vt16 = (_Float16*)(ws + 3 * wt_bytes + 2 * qk_bytes);

    const float qscale = 0.125f;  // 1/sqrt(H), folded into Wq
    const int wt_threads = 256;
    const int wt_blocks  = (DD * HH + wt_threads - 1) / wt_threads;
    wt_convert_kernel<<<wt_blocks, wt_threads, 0, stream>>>(Wq, Wtq, qscale);
    wt_convert_kernel<<<wt_blocks, wt_threads, 0, stream>>>(Wk, Wtk, 1.0f);
    wt_convert_kernel<<<wt_blocks, wt_threads, 0, stream>>>(Wv, Wtv, 1.0f);

    const int nblocks = BB * (TT / 16);  // 1024 waves
    qkv_proj_kernel<<<nblocks, 32, 0, stream>>>(x, Wtq, Wtk, Wtv, Q16, K16, Vt16);
    flash_attn_kernel<<<nblocks, 32, 0, stream>>>(Q16, K16, Vt16, out);
}
